// HAGCNBlock_31078383354251
// MI455X (gfx1250) — matrix operand order repaired and checked by reading, not compiled, not run on hardware
//
#include <hip/hip_runtime.h>
#include <hip/hip_bf16.h>

// ---------------------------------------------------------------------------
// HAGCN block for MI455X (gfx1250): bf16 WMMA (v_wmma_f32_16x16x32_bf16),
// fp32 accumulate, LDS-staged transposed activations, bf16 intermediate y1.
// ---------------------------------------------------------------------------

typedef __attribute__((ext_vector_type(16))) __bf16 v16bf;
typedef __attribute__((ext_vector_type(8)))  __bf16 v8bf;
typedef __attribute__((ext_vector_type(8)))  float  v8f;

#define C_CH   256
#define T_LEN  256
#define V_J    27
#define V_PAD  32
#define K_SUB  3
#define N_B    32

static __device__ __forceinline__ v16bf cat8(v8bf lo, v8bf hi) {
    return __builtin_shufflevector(lo, hi, 0,1,2,3,4,5,6,7,8,9,10,11,12,13,14,15);
}

// A-matrix fragment, 16x32 bf16, row-major source with row stride `ld`.
// ISA layout: lanes 0-15 hold row=lane, K=0..7 (V0-3) and K=16..23 (V4-7);
// lanes 16-31 hold row=lane-16, K=8..15 and K=24..31.
static __device__ __forceinline__ v16bf load_a(const __bf16* __restrict__ tile,
                                               int ld, int ks, int lane) {
    const int ca = (lane < 16) ? 0 : 8;
    const __bf16* p = tile + (lane & 15) * ld + ks + ca;
    return cat8(*(const v8bf*)p, *(const v8bf*)(p + 16));
}

// B-matrix fragment, 32x16 bf16, from a K-contiguous (transposed) buffer with
// row stride `ld` (row index = N column). ISA layout: lanes 0-15 hold col=lane,
// K=0..7 and K=16..23; lanes 16-31 hold col=lane-16, K=8..15 and K=24..31.
static __device__ __forceinline__ v16bf load_b(const __bf16* __restrict__ tcol,
                                               int col, int ld, int ks, int lane) {
    const __bf16* p = tcol + col * ld + ks + ((lane < 16) ? 0 : 8);
    return cat8(*(const v8bf*)p, *(const v8bf*)(p + 16));
}

static __device__ __forceinline__ v8f wmma_bf16(v16bf a, v16bf b, v8f c) {
    return __builtin_amdgcn_wmma_f32_16x16x32_bf16(false, a, false, b,
                                                   (short)0, c, false, false);
}

// ---------------------------------------------------------------------------
// Prep kernels: weight fp32 -> bf16 repacks, BN coefficient folding.
// ---------------------------------------------------------------------------
__global__ void prep_wg(const float* __restrict__ Wg, unsigned short* __restrict__ WgB_u) {
    __bf16* WgB = (__bf16*)WgB_u;
    int i = blockIdx.x * blockDim.x + threadIdx.x;
    if (i < K_SUB * C_CH * C_CH) WgB[i] = (__bf16)Wg[i];
}

__global__ void prep_wt(const float* __restrict__ Wt, unsigned short* __restrict__ WtB_u) {
    // Wt layout (O,C,9,1) -> WtB[dt][o][c] row-major
    __bf16* WtB = (__bf16*)WtB_u;
    int i = blockIdx.x * blockDim.x + threadIdx.x;
    if (i < 9 * C_CH * C_CH) {
        int dt = i >> 16;          // / 65536
        int r  = i & 65535;
        int o  = r >> 8;
        int c  = r & 255;
        WtB[i] = (__bf16)Wt[(o * C_CH + c) * 9 + dt];
    }
}

__global__ void prep_bn(const float* __restrict__ g1g, const float* __restrict__ g1b,
                        const float* __restrict__ g1m, const float* __restrict__ g1v,
                        const float* __restrict__ g2g, const float* __restrict__ g2b,
                        const float* __restrict__ g2m, const float* __restrict__ g2v,
                        const float* __restrict__ bt,
                        float* __restrict__ s1, float* __restrict__ b1,
                        float* __restrict__ s2, float* __restrict__ b2p) {
    int o = threadIdx.x;
    if (o < C_CH) {
        float sa = g1g[o] * rsqrtf(g1v[o] + 1e-5f);
        s1[o] = sa;
        b1[o] = g1b[o] - g1m[o] * sa;
        float sb = g2g[o] * rsqrtf(g2v[o] + 1e-5f);
        s2[o]  = sb;
        b2p[o] = g2b[o] - g2m[o] * sb + sb * bt[o];   // bt folded into BN2 bias
    }
}

// ---------------------------------------------------------------------------
// Kernel 1: HandAwareGCN. One workgroup (8 waves) per (n,t).
//   f_k(256x32) = Wg_k @ X ; y += (f_k + bg_k) @ A_comb_k ; BN1 + res + relu.
// Emits y1 as bf16 (N,C,T,V) into workspace.
// ---------------------------------------------------------------------------
__global__ __launch_bounds__(256) void hagcn_gcn(
    const float* __restrict__ x,
    const float* __restrict__ A,  const float* __restrict__ SH,
    const float* __restrict__ PH, const float* __restrict__ PA,
    const float* __restrict__ alpha, const float* __restrict__ beta,
    const unsigned short* __restrict__ WgB_u,
    const float* __restrict__ bg,
    const float* __restrict__ s1, const float* __restrict__ b1,
    unsigned short* __restrict__ y1B_u)
{
    __shared__ __bf16 XT[V_PAD * C_CH];      // [v][c] transposed activations
    __shared__ __bf16 fS[C_CH * V_PAD];      // [m][v] spilled f_k (bf16)
    __shared__ __bf16 AkT[K_SUB * V_PAD * V_PAD]; // [k][w][v] transposed A_comb

    const __bf16* WgB = (const __bf16*)WgB_u;
    __bf16* y1B = (__bf16*)y1B_u;

    const int tid  = threadIdx.x;
    const int lane = tid & 31;
    const int wv   = tid >> 5;           // 0..7
    const int blk  = blockIdx.x;
    const int n    = blk >> 8;
    const int t    = blk & 255;

    // ---- stage X^T (fp32 -> bf16), zero V padding rows ----
    {
        const float* xp = x + ((size_t)(n * C_CH + tid) * T_LEN + t) * V_J;
        #pragma unroll
        for (int v = 0; v < V_J; ++v)  XT[v * C_CH + tid] = (__bf16)xp[v];
        #pragma unroll
        for (int v = V_J; v < V_PAD; ++v) XT[v * C_CH + tid] = (__bf16)0.0f;
    }
    // ---- stage A_comb^T = (A + PA + SH*alpha + PH*beta)^T, zero-padded ----
    {
        const float al = alpha[0], be = beta[0];
        for (int j = tid; j < K_SUB * V_PAD * V_PAD; j += 256) {
            int k = j >> 10, r = j & 1023, w = r >> 5, v = r & 31;
            float val = 0.0f;
            if (v < V_J && w < V_J) {
                int idx = (k * V_J + v) * V_J + w;
                val = A[idx] + PA[idx] + SH[idx] * al + PH[idx] * be;
            }
            AkT[(k * V_PAD + w) * V_PAD + v] = (__bf16)val;
        }
    }
    __syncthreads();

    const int mrow0 = wv * 32;           // wave owns rows mrow0..mrow0+31
    v8f yacc[2][2] = {};

    for (int k = 0; k < K_SUB; ++k) {
        v8f facc[2][2] = {};
        const __bf16* wg = WgB + k * (C_CH * C_CH);

        // f_k = Wg_k @ X   (M=256 per-wave 32, N=32, K=256)
        for (int ks = 0; ks < C_CH; ks += 32) {
            #pragma unroll
            for (int nt = 0; nt < 2; ++nt) {
                v16bf bf = load_b(XT, (lane & 15) + nt * 16, C_CH, ks, lane);
                #pragma unroll
                for (int mt = 0; mt < 2; ++mt) {
                    v16bf af = load_a(wg + (mrow0 + mt * 16) * C_CH, C_CH, ks, lane);
                    facc[mt][nt] = wmma_bf16(af, bf, facc[mt][nt]);
                }
            }
        }

        // spill f_k (+bg_k) to LDS as bf16 A-operand
        __syncthreads();
        #pragma unroll
        for (int mt = 0; mt < 2; ++mt)
            #pragma unroll
            for (int nt = 0; nt < 2; ++nt)
                #pragma unroll
                for (int r = 0; r < 8; ++r) {
                    int m = mrow0 + mt * 16 + r + ((lane >> 4) << 3);
                    int v = nt * 16 + (lane & 15);
                    fS[m * V_PAD + v] = (__bf16)(facc[mt][nt][r] + bg[k * C_CH + m]);
                }
        __syncthreads();

        // y += f_k @ A_comb_k   (K = V_PAD = 32, single WMMA step)
        const __bf16* akt = AkT + k * (V_PAD * V_PAD);
        #pragma unroll
        for (int nt = 0; nt < 2; ++nt) {
            v16bf bf = load_b(akt, (lane & 15) + nt * 16, V_PAD, 0, lane);
            #pragma unroll
            for (int mt = 0; mt < 2; ++mt) {
                v16bf af = load_a(fS + (mrow0 + mt * 16) * V_PAD, V_PAD, 0, lane);
                yacc[mt][nt] = wmma_bf16(af, bf, yacc[mt][nt]);
            }
        }
    }

    // ---- epilogue: BN1 + residual + relu -> bf16 y1 ----
    #pragma unroll
    for (int mt = 0; mt < 2; ++mt)
        #pragma unroll
        for (int nt = 0; nt < 2; ++nt)
            #pragma unroll
            for (int r = 0; r < 8; ++r) {
                int m = mrow0 + mt * 16 + r + ((lane >> 4) << 3);
                int v = nt * 16 + (lane & 15);
                if (v < V_J) {
                    size_t xi = ((size_t)(n * C_CH + m) * T_LEN + t) * V_J + v;
                    float val = yacc[mt][nt][r] * s1[m] + b1[m] + x[xi];
                    y1B[xi] = (__bf16)(val > 0.0f ? val : 0.0f);
                }
            }
}

// ---------------------------------------------------------------------------
// Kernel 2: TemporalConv (9,1) + BN2 + residual + relu. One workgroup per (n,t).
//   out = relu( s2*conv(y1) + (s2*bt + b2) + x )
// y1 (bf16, 113 MB) is ~L2-resident, so the 9x temporal reuse is cheap.
// ---------------------------------------------------------------------------
__global__ __launch_bounds__(256) void hagcn_tconv(
    const float* __restrict__ x,
    const unsigned short* __restrict__ y1B_u,
    const unsigned short* __restrict__ WtB_u,
    const float* __restrict__ s2, const float* __restrict__ b2p,
    float* __restrict__ out)
{
    __shared__ __bf16 XT[V_PAD * C_CH];

    const __bf16* y1B = (const __bf16*)y1B_u;
    const __bf16* WtB = (const __bf16*)WtB_u;

    const int tid  = threadIdx.x;
    const int lane = tid & 31;
    const int wv   = tid >> 5;
    const int blk  = blockIdx.x;
    const int n    = blk >> 8;
    const int t    = blk & 255;
    const int mrow0 = wv * 32;

    v8f yacc[2][2] = {};

    for (int dt = 0; dt < 9; ++dt) {
        const int ts = t + dt - 4;
        if (ts < 0 || ts >= T_LEN) continue;       // block-uniform

        __syncthreads();                            // previous dt's reads done
        {
            const __bf16* yp = y1B + ((size_t)(n * C_CH + tid) * T_LEN + ts) * V_J;
            #pragma unroll
            for (int v = 0; v < V_J; ++v)  XT[v * C_CH + tid] = yp[v];
            #pragma unroll
            for (int v = V_J; v < V_PAD; ++v) XT[v * C_CH + tid] = (__bf16)0.0f;
            if (ts + 1 < T_LEN)                     // hint: next temporal slice
                __builtin_prefetch(yp + V_J, 0, 1);
        }
        __syncthreads();

        const __bf16* wt = WtB + dt * (C_CH * C_CH);
        for (int ks = 0; ks < C_CH; ks += 32) {
            #pragma unroll
            for (int nt = 0; nt < 2; ++nt) {
                v16bf bf = load_b(XT, (lane & 15) + nt * 16, C_CH, ks, lane);
                #pragma unroll
                for (int mt = 0; mt < 2; ++mt) {
                    v16bf af = load_a(wt + (mrow0 + mt * 16) * C_CH, C_CH, ks, lane);
                    yacc[mt][nt] = wmma_bf16(af, bf, yacc[mt][nt]);
                }
            }
        }
    }

    // ---- epilogue: BN2(+bt) + residual + relu -> fp32 out ----
    #pragma unroll
    for (int mt = 0; mt < 2; ++mt)
        #pragma unroll
        for (int nt = 0; nt < 2; ++nt)
            #pragma unroll
            for (int r = 0; r < 8; ++r) {
                int m = mrow0 + mt * 16 + r + ((lane >> 4) << 3);
                int v = nt * 16 + (lane & 15);
                if (v < V_J) {
                    size_t xi = ((size_t)(n * C_CH + m) * T_LEN + t) * V_J + v;
                    float val = yacc[mt][nt][r] * s2[m] + b2p[m] + x[xi];
                    out[xi] = val > 0.0f ? val : 0.0f;
                }
            }
}

// ---------------------------------------------------------------------------
// Workspace layout (bytes):
//   WgB  : 3*256*256*2      =    393,216
//   WtB  : 9*256*256*2      =  1,179,648
//   s1/b1/s2/b2p            =      4,096
//   y1B  : 32*256*256*27*2  = 113,246,208
// total ~114.8 MB
// ---------------------------------------------------------------------------
extern "C" void kernel_launch(void* const* d_in, const int* in_sizes, int n_in,
                              void* d_out, int out_size, void* d_ws, size_t ws_size,
                              hipStream_t stream) {
    const float* x     = (const float*)d_in[0];
    const float* A     = (const float*)d_in[1];
    const float* SH    = (const float*)d_in[2];
    const float* PH    = (const float*)d_in[3];
    const float* PA    = (const float*)d_in[4];
    const float* alpha = (const float*)d_in[5];
    const float* beta  = (const float*)d_in[6];
    const float* Wg    = (const float*)d_in[7];
    const float* bg    = (const float*)d_in[8];
    const float* g1g   = (const float*)d_in[9];
    const float* g1b   = (const float*)d_in[10];
    const float* g1m   = (const float*)d_in[11];
    const float* g1v   = (const float*)d_in[12];
    const float* Wt    = (const float*)d_in[13];
    const float* bt    = (const float*)d_in[14];
    const float* g2g   = (const float*)d_in[15];
    const float* g2b   = (const float*)d_in[16];
    const float* g2m   = (const float*)d_in[17];
    const float* g2v   = (const float*)d_in[18];
    // d_in[19] = keep_prob == 1 -> identity

    char* ws = (char*)d_ws;
    unsigned short* WgB = (unsigned short*)ws;
    unsigned short* WtB = (unsigned short*)(ws + 393216);
    float* s1  = (float*)(ws + 393216 + 1179648);
    float* b1  = s1 + 256;
    float* s2  = b1 + 256;
    float* b2p = s2 + 256;
    unsigned short* y1B = (unsigned short*)(ws + 393216 + 1179648 + 4096);

    prep_wg<<<(K_SUB * C_CH * C_CH + 255) / 256, 256, 0, stream>>>(Wg, WgB);
    prep_wt<<<(9 * C_CH * C_CH + 255) / 256, 256, 0, stream>>>(Wt, WtB);
    prep_bn<<<1, 256, 0, stream>>>(g1g, g1b, g1m, g1v, g2g, g2b, g2m, g2v, bt,
                                   s1, b1, s2, b2p);

    const int blocks = N_B * T_LEN;   // 8192
    hagcn_gcn<<<blocks, 256, 0, stream>>>(x, A, SH, PH, PA, alpha, beta,
                                          WgB, bg, s1, b1, y1B);
    hagcn_tconv<<<blocks, 256, 0, stream>>>(x, y1B, WtB, s2, b2p, (float*)d_out);
}
